// LSTMClassifier_77936476553600
// MI455X (gfx1250) — matrix operand order repaired and checked by reading, not compiled, not run on hardware
//
#include <hip/hip_runtime.h>
#include <math.h>

// ---------------------------------------------------------------------------
// LSTMClassifier for MI455X (gfx1250, wave32, WMMA).
//   B=512 T=200 D=64 H=256 ; C1=1024 C2=256 C3=10 R1=1024
// All GEMMs on v_wmma_f32_16x16x32_f16, f32 accumulation.
// f32 -> f16 conversion fused into LDS staging; double-buffered LDS pipeline.
// Compile-time K so all addressing is shifts/constants (no v_mul_u64).
// ---------------------------------------------------------------------------

typedef _Float16 v4h  __attribute__((ext_vector_type(4)));
typedef _Float16 v8h  __attribute__((ext_vector_type(8)));
typedef _Float16 v16h __attribute__((ext_vector_type(16)));
typedef float    v8f  __attribute__((ext_vector_type(8)));

__device__ __forceinline__ v16h cat16(v8h lo, v8h hi) {
    return __builtin_shufflevector(lo, hi, 0,1,2,3,4,5,6,7,8,9,10,11,12,13,14,15);
}

__device__ __forceinline__ v4h cvt4(float4 v) {
    return v4h{(_Float16)v.x, (_Float16)v.y, (_Float16)v.z, (_Float16)v.w};
}

// A operand fragment (16x32 f16, MxK). LDS tile: 16 rows x 32 halfs row-major.
// lane<16: m=lane, K=0..7 then 16..23 ; lane>=16: m=lane-16, K=8..15 then 24..31.
__device__ __forceinline__ v16h load_a_frag(const _Float16* base, int lane) {
    const int m  = lane & 15;
    const int h8 = lane >> 4;
    const v8h lo = *reinterpret_cast<const v8h*>(base + m * 32 + h8 * 8);
    const v8h hi = *reinterpret_cast<const v8h*>(base + m * 32 + h8 * 8 + 16);
    return cat16(lo, hi);
}

// B operand fragment (32x16 f16, KxN). LDS tile: 16 rows (n) x 32 halfs (k).
// lane<16: n=lane, K=0..7 then 16..23 ; lane>=16: n=lane-16, K=8..15 then 24..31.
__device__ __forceinline__ v16h load_b_frag(const _Float16* base, int lane) {
    const int n  = lane & 15;
    const int kh = lane >> 4;
    const v8h lo = *reinterpret_cast<const v8h*>(base + n * 32 + kh * 8);
    const v8h hi = *reinterpret_cast<const v8h*>(base + n * 32 + kh * 8 + 16);
    return cat16(lo, hi);
}

__device__ __forceinline__ v8f wmma_f16(v16h a, v16h b, v8f c) {
    return __builtin_amdgcn_wmma_f32_16x16x32_f16(false, a, false, b, (short)0, c, false, false);
}

__device__ __forceinline__ float sigmoidf_(float x) {
    return 1.0f / (1.0f + __expf(-x));
}

// ---- register staging (uniform per-thread assignments, compile-time K) ----
// A tile: 128 rows x 32 floats = 1024 float4, 4 per thread (half row each).
template <int K>
__device__ __forceinline__ void loadA(const float* __restrict__ A, int mBase,
                                      int k0, int tid, float4 (&r)[4]) {
    const int row = tid >> 1;
    const int c0  = (tid & 1) * 16;
    const float* p = A + (size_t)(mBase + row) * K + k0 + c0;
#pragma unroll
    for (int i = 0; i < 4; ++i) r[i] = *reinterpret_cast<const float4*>(p + i * 4);
    __builtin_prefetch(p + 64, 0, 0);   // two K-tiles ahead -> global_prefetch_b8
}
__device__ __forceinline__ void storeA(_Float16* sA, int tid, const float4 (&r)[4]) {
    const int row = tid >> 1;
    const int c0  = (tid & 1) * 16;
#pragma unroll
    for (int i = 0; i < 4; ++i)
        *reinterpret_cast<v4h*>(&sA[row * 32 + c0 + i * 4]) = cvt4(r[i]);
}

// B tile: 64 rows x 32 floats = 512 float4, 2 per thread. Rows >= N zero-filled.
template <int K>
__device__ __forceinline__ void loadB(const float* __restrict__ B, int nBase, int N,
                                      int k0, int tid, float4 (&r)[2]) {
#pragma unroll
    for (int i = 0; i < 2; ++i) {
        const int idx = tid * 2 + i;
        const int row = idx >> 3, c = (idx & 7) * 4;
        const int gn  = nBase + row;
        r[i] = (gn < N) ? *reinterpret_cast<const float4*>(B + (size_t)gn * K + k0 + c)
                        : make_float4(0.f, 0.f, 0.f, 0.f);
    }
}
// LSTM weight tile: 4 gates x 16 h-rows (row = gate*16 + j), always in range.
template <int K>
__device__ __forceinline__ void loadBg(const float* __restrict__ W, int hBase,
                                       int k0, int tid, float4 (&r)[2]) {
#pragma unroll
    for (int i = 0; i < 2; ++i) {
        const int idx = tid * 2 + i;
        const int row = idx >> 3, c = (idx & 7) * 4;
        const int gn  = (row >> 4) * 256 + hBase + (row & 15);
        r[i] = *reinterpret_cast<const float4*>(W + (size_t)gn * K + k0 + c);
    }
}
__device__ __forceinline__ void storeB(_Float16* sB, int tid, const float4 (&r)[2]) {
#pragma unroll
    for (int i = 0; i < 2; ++i) {
        const int idx = tid * 2 + i;
        const int row = idx >> 3, c = (idx & 7) * 4;
        *reinterpret_cast<v4h*>(&sB[row * 32 + c]) = cvt4(r[i]);
    }
}

// ---------------------------------------------------------------------------
// Fused LSTM step from zero state: out[m,h] = sig(o)*tanh(sig(i)*tanh(g))
// gates = A[M,K] @ W[4H,K]^T + b_ih + b_hh ; forget gate never multiplied.
// Block 256 thr (8 waves); tile 128M x 16h; grid (M/128, H/16).
// ---------------------------------------------------------------------------
template <int K>
__global__ __launch_bounds__(256)
void lstm_gemm_kernel(const float* __restrict__ A, const float* __restrict__ W,
                      const float* __restrict__ b_ih, const float* __restrict__ b_hh,
                      float* __restrict__ out) {
    constexpr int H  = 256;
    constexpr int NT = K / 32;
    __shared__ __align__(16) _Float16 sA[2][128 * 32];   // 16 KB
    __shared__ __align__(16) _Float16 sB[2][64 * 32];    // 8 KB

    const int tid   = threadIdx.x;
    const int lane  = tid & 31;
    const int wave  = tid >> 5;
    const int mBase = blockIdx.x * 128;
    const int hBase = blockIdx.y * 16;

    v8f acc_i = {}, acc_g = {}, acc_o = {};
    float4 ra[4], rb[2];

    loadA<K>(A, mBase, 0, tid, ra);
    loadBg<K>(W, hBase, 0, tid, rb);
    storeA(sA[0], tid, ra);
    storeB(sB[0], tid, rb);
    __syncthreads();

#pragma unroll 2
    for (int kt = 0; kt < NT; ++kt) {
        if (kt + 1 < NT) {                       // next tile's loads in flight
            loadA<K>(A, mBase, (kt + 1) * 32, tid, ra);
            loadBg<K>(W, hBase, (kt + 1) * 32, tid, rb);
        }
        const _Float16* cA = sA[kt & 1];
        const _Float16* cB = sB[kt & 1];
        const v16h a  = load_a_frag(cA + wave * 16 * 32, lane);
        const v16h bi = load_b_frag(cB + 0 * 16 * 32, lane);   // gate i
        const v16h bg = load_b_frag(cB + 2 * 16 * 32, lane);   // gate g
        const v16h bo = load_b_frag(cB + 3 * 16 * 32, lane);   // gate o
        acc_i = wmma_f16(a, bi, acc_i);
        acc_g = wmma_f16(a, bg, acc_g);
        acc_o = wmma_f16(a, bo, acc_o);
        if (kt + 1 < NT) {
            storeA(sA[(kt + 1) & 1], tid, ra);
            storeB(sB[(kt + 1) & 1], tid, rb);
        }
        __syncthreads();
    }

    const int nl   = lane & 15;
    const int hcol = hBase + nl;
    const float bbi = b_ih[hcol]         + b_hh[hcol];
    const float bbg = b_ih[2 * H + hcol] + b_hh[2 * H + hcol];
    const float bbo = b_ih[3 * H + hcol] + b_hh[3 * H + hcol];
    const int mOff = mBase + 16 * wave + ((lane >> 4) << 3);
#pragma unroll
    for (int r = 0; r < 8; ++r) {
        const float gi = acc_i[r] + bbi;
        const float gg = acc_g[r] + bbg;
        const float go = acc_o[r] + bbo;
        const float c  = sigmoidf_(gi) * tanhf(gg);
        out[(size_t)(mOff + r) * H + hcol] = sigmoidf_(go) * tanhf(c);
    }
}

// ---------------------------------------------------------------------------
// Head GEMM: C = act(A[M,K] @ B[N,K]^T + bias). Tile 128M x 64N, 8 waves.
// SPLIT: writes raw partials at out + z*M*N (bias/act in reduce kernel).
// ---------------------------------------------------------------------------
template <int ACT, bool SPLIT, int K, int KCHUNK>
__global__ __launch_bounds__(256)
void gemm_head_kernel(const float* __restrict__ A, const float* __restrict__ B,
                      const float* __restrict__ bias, float* __restrict__ out,
                      int M, int N) {
    constexpr int NT = KCHUNK / 32;
    __shared__ __align__(16) _Float16 sA[2][128 * 32];   // 16 KB
    __shared__ __align__(16) _Float16 sB[2][64 * 32];    // 8 KB

    const int tid    = threadIdx.x;
    const int lane   = tid & 31;
    const int wave   = tid >> 5;
    const int mBase  = blockIdx.x * 128;
    const int nBase  = blockIdx.y * 64;
    const int kStart = blockIdx.z * KCHUNK;

    v8f acc[4] = {{}, {}, {}, {}};
    float4 ra[4], rb[2];

    loadA<K>(A, mBase, kStart, tid, ra);
    loadB<K>(B, nBase, N, kStart, tid, rb);
    storeA(sA[0], tid, ra);
    storeB(sB[0], tid, rb);
    __syncthreads();

#pragma unroll 2
    for (int kt = 0; kt < NT; ++kt) {
        if (kt + 1 < NT) {
            loadA<K>(A, mBase, kStart + (kt + 1) * 32, tid, ra);
            loadB<K>(B, nBase, N, kStart + (kt + 1) * 32, tid, rb);
        }
        const _Float16* cA = sA[kt & 1];
        const _Float16* cB = sB[kt & 1];
        const v16h a = load_a_frag(cA + wave * 16 * 32, lane);
#pragma unroll
        for (int nt = 0; nt < 4; ++nt) {
            const v16h b = load_b_frag(cB + nt * 16 * 32, lane);
            acc[nt] = wmma_f16(a, b, acc[nt]);
        }
        if (kt + 1 < NT) {
            storeA(sA[(kt + 1) & 1], tid, ra);
            storeB(sB[(kt + 1) & 1], tid, rb);
        }
        __syncthreads();
    }

    float* o = out;
    if (SPLIT) o += (size_t)blockIdx.z * (size_t)M * (size_t)N;
    const int mOff = mBase + 16 * wave + ((lane >> 4) << 3);
#pragma unroll
    for (int nt = 0; nt < 4; ++nt) {
        const int n = nBase + nt * 16 + (lane & 15);
        if (n >= N) continue;
        const float bv = (!SPLIT && bias) ? bias[n] : 0.0f;
#pragma unroll
        for (int r = 0; r < 8; ++r) {
            float v = acc[nt][r];
            if (!SPLIT) {
                v += bv;
                if (ACT == 1) v = fmaxf(v, 0.0f);
            }
            o[(size_t)(mOff + r) * N + n] = v;
        }
    }
}

// ---------------------------------------------------------------------------
// Split-K reduction: out[m,n] = relu(bias[n] + sum_z partial[z][m][n])
// ---------------------------------------------------------------------------
__global__ __launch_bounds__(256)
void reduce_bias_relu_kernel(const float* __restrict__ partial,
                             const float* __restrict__ bias,
                             float* __restrict__ out, int MN, int N, int Z) {
    const int idx = blockIdx.x * 256 + threadIdx.x;
    if (idx >= MN) return;
    float s = 0.0f;
    for (int z = 0; z < Z; ++z) s += partial[(size_t)z * MN + idx];
    s += bias[idx % N];
    out[idx] = fmaxf(s, 0.0f);
}

// ---------------------------------------------------------------------------
extern "C" void kernel_launch(void* const* d_in, const int* in_sizes, int n_in,
                              void* d_out, int out_size, void* d_ws, size_t ws_size,
                              hipStream_t stream) {
    (void)in_sizes; (void)n_in; (void)out_size; (void)ws_size;

    const float* x     = (const float*)d_in[0];
    const float* W_ih0 = (const float*)d_in[1];
    const float* b_ih0 = (const float*)d_in[2];
    const float* b_hh0 = (const float*)d_in[3];
    const float* W_ih1 = (const float*)d_in[4];
    const float* b_ih1 = (const float*)d_in[5];
    const float* b_hh1 = (const float*)d_in[6];
    const float* fc1_w = (const float*)d_in[7];
    const float* fc1_b = (const float*)d_in[8];
    const float* fc2_w = (const float*)d_in[9];
    const float* fc2_b = (const float*)d_in[10];
    const float* fc3_w = (const float*)d_in[11];
    const float* fc3_b = (const float*)d_in[12];
    const float* fc4_w = (const float*)d_in[13];
    const float* fc4_b = (const float*)d_in[14];
    const float* fc5_w = (const float*)d_in[15];
    const float* fc5_b = (const float*)d_in[16];

    constexpr int Bz = 512, T = 200, D = 64, H = 256;
    constexpr int M0 = Bz * T;            // 102400
    constexpr int KF = H * T;             // 51200
    constexpr int C1 = 1024, C2 = 256, C3 = 10, R1 = 1024;
    constexpr int KSPLIT = 25;            // 51200/25 = 2048 = 64 k-iters per block
    constexpr int KCHUNK = KF / KSPLIT;

    // d_out (tuple, flat): out_total[512,200,256] | class[512,10] | reg[512,200]
    float* out_total = (float*)d_out;
    float* out_class = out_total + (size_t)M0 * H;
    float* out_reg   = out_class + (size_t)Bz * C3;
    const float* flat = out_total;        // [512, 51200] view of out_total

    // ws (floats): h0 (reused as split-K partials after G1) | c1 | r | c2
    float* ws      = (float*)d_ws;
    float* h0      = ws;                                  // [102400,256]
    float* partial = ws;                                  // [25,512,1024]
    float* c1      = ws + (size_t)M0 * H;                 // [512,1024]
    float* rr      = c1 + (size_t)Bz * C1;                // [512,1024]
    float* c2      = rr + (size_t)Bz * R1;                // [512,256]

    const dim3 blk(256);

    // LSTM layer 0: x[102400,64] -> h0[102400,256]
    lstm_gemm_kernel<D><<<dim3(M0 / 128, H / 16), blk, 0, stream>>>(
        x, W_ih0, b_ih0, b_hh0, h0);
    // LSTM layer 1: h0 -> out_total (== flat)
    lstm_gemm_kernel<H><<<dim3(M0 / 128, H / 16), blk, 0, stream>>>(
        h0, W_ih1, b_ih1, b_hh1, out_total);

    // fc1: flat @ fc1_w^T -> c1 (split-K partials + reduce with bias+relu)
    gemm_head_kernel<0, true, KF, KCHUNK>
        <<<dim3(Bz / 128, C1 / 64, KSPLIT), blk, 0, stream>>>(
            flat, fc1_w, nullptr, partial, Bz, C1);
    reduce_bias_relu_kernel<<<(Bz * C1 + 255) / 256, blk, 0, stream>>>(
        partial, fc1_b, c1, Bz * C1, C1, KSPLIT);

    // fc2: relu(c1 @ fc2_w^T + b) -> c2
    gemm_head_kernel<1, false, C1, C1><<<dim3(Bz / 128, C2 / 64, 1), blk, 0, stream>>>(
        c1, fc2_w, fc2_b, c2, Bz, C2);

    // fc3: c2 @ fc3_w^T + b -> class logits (N=10, masked)
    gemm_head_kernel<0, false, C2, C2><<<dim3(Bz / 128, 1, 1), blk, 0, stream>>>(
        c2, fc3_w, fc3_b, out_class, Bz, C3);

    // fc4: flat @ fc4_w^T -> rr (split-K partials + reduce with bias+relu)
    gemm_head_kernel<0, true, KF, KCHUNK>
        <<<dim3(Bz / 128, R1 / 64, KSPLIT), blk, 0, stream>>>(
            flat, fc4_w, nullptr, partial, Bz, R1);
    reduce_bias_relu_kernel<<<(Bz * R1 + 255) / 256, blk, 0, stream>>>(
        partial, fc4_b, rr, Bz * R1, R1, KSPLIT);

    // fc5: rr @ fc5_w^T + b -> reg output (N=200, masked)
    gemm_head_kernel<0, false, R1, R1>
        <<<dim3(Bz / 128, (T + 63) / 64, 1), blk, 0, stream>>>(
            rr, fc5_w, fc5_b, out_reg, Bz, T);
}
